// EpisodeMultiheadAttentionBlock_81226421502607
// MI455X (gfx1250) — matrix operand order repaired and checked by reading, not compiled, not run on hardware
//
#include <hip/hip_runtime.h>

// EpisodeMultiheadAttentionBlock for MI455X (gfx1250, wave32, WMMA bf16).
// bf16 WMMA GEMMs with fused epilogues (RoPE / transpose / sigmoid / tanh-gate),
// LDS-tiled attention with mean-over-heads written directly to d_out.
// This round: software-pipelined K-loops (prefetch next fragments before the
// current WMMAs) so waits are partial instead of s_wait_loadcnt 0 before every
// v_wmma, letting load clauses overlap the matrix pipe.

typedef unsigned short u16;
typedef __attribute__((ext_vector_type(16))) __bf16 v16bf;
typedef __attribute__((ext_vector_type(8)))  float  v8f;
typedef __attribute__((ext_vector_type(4)))  unsigned int u32x4;
typedef __attribute__((ext_vector_type(8)))  unsigned int u32x8;

#define BATCH 16
#define SEQ_K 1024
#define SEQ_Q 512
#define EMBED 1024
#define HEADS 16
#define HDIM  64

static __device__ __forceinline__ u16 f2bf(float x) {
  unsigned u = __builtin_bit_cast(unsigned, x);
  u += 0x7FFFu + ((u >> 16) & 1u);  // round-to-nearest-even
  return (u16)(u >> 16);
}
static __device__ __forceinline__ float bf2f(u16 h) {
  unsigned u = ((unsigned)h) << 16;
  return __builtin_bit_cast(float, u);
}

// Combine two 16B chunks into a 32B bf16 fragment without touching memory.
static __device__ __forceinline__ v16bf frag_pack(u32x4 a, u32x4 b) {
  u32x8 c;
#pragma unroll
  for (int i = 0; i < 4; ++i) { c[i] = a[i]; c[i + 4] = b[i]; }
  return __builtin_bit_cast(v16bf, c);
}
// A fragment (16x32 bf16, MxK): lane l<16 -> row l, K chunks at hi*8 and hi*8+16.
static __device__ __forceinline__ v16bf load_frag_a(const u16* __restrict__ base,
                                                    int row0, int stride, int k0,
                                                    int lane) {
  const int l = lane & 15, hi = lane >> 4;
  const u16* p = base + (size_t)(row0 + l) * stride + (k0 + hi * 8);
  return frag_pack(*(const u32x4*)(p), *(const u32x4*)(p + 16));
}
// B fragment (32x16 bf16, KxN) from [N][K]-major: lane l -> col N=l, K chunks
// at k0 + hi*8 and k0 + hi*8 + 16 (matches A fragment).
static __device__ __forceinline__ v16bf load_frag_b(const u16* __restrict__ base,
                                                    int row0, int stride, int k0,
                                                    int lane) {
  const int l = lane & 15, hi = lane >> 4;
  const u16* p = base + (size_t)(row0 + l) * stride + (k0 + hi * 8);
  return frag_pack(*(const u32x4*)(p), *(const u32x4*)(p + 16));
}
static __device__ __forceinline__ v8f wmma_bf16(v16bf a, v16bf b, v8f c) {
  return __builtin_amdgcn_wmma_f32_16x16x32_bf16(false, a, false, b, (short)0, c,
                                                 false, false);
}
static __device__ __forceinline__ v8f zero8() {
  v8f z;
#pragma unroll
  for (int i = 0; i < 8; ++i) z[i] = 0.0f;
  return z;
}
static __device__ __forceinline__ float xquery(const float* __restrict__ key_in,
                                               int m, int n) {
  const int b = m >> 9, s = m & 511;
  return key_in[((size_t)(b * SEQ_K + SEQ_Q + s) << 10) + n];
}

// One K=1024 pass of acc[0..3] += A[16 x K] @ W[K x 64], software-pipelined.
static __device__ __forceinline__ void gemm_pass(const u16* __restrict__ A,
                                                 const u16* __restrict__ WT,
                                                 v8f (&acc)[4], int mbase, int nbase,
                                                 int lane) {
  v16bf a = load_frag_a(A, mbase, EMBED, 0, lane);
  v16bf b0 = load_frag_b(WT, nbase + 0, EMBED, 0, lane);
  v16bf b1 = load_frag_b(WT, nbase + 16, EMBED, 0, lane);
  v16bf b2 = load_frag_b(WT, nbase + 32, EMBED, 0, lane);
  v16bf b3 = load_frag_b(WT, nbase + 48, EMBED, 0, lane);
  for (int kc = 0; kc < EMBED - 32; kc += 32) {
    v16bf an = load_frag_a(A, mbase, EMBED, kc + 32, lane);
    v16bf c0 = load_frag_b(WT, nbase + 0, EMBED, kc + 32, lane);
    v16bf c1 = load_frag_b(WT, nbase + 16, EMBED, kc + 32, lane);
    v16bf c2 = load_frag_b(WT, nbase + 32, EMBED, kc + 32, lane);
    v16bf c3 = load_frag_b(WT, nbase + 48, EMBED, kc + 32, lane);
    acc[0] = wmma_bf16(a, b0, acc[0]);
    acc[1] = wmma_bf16(a, b1, acc[1]);
    acc[2] = wmma_bf16(a, b2, acc[2]);
    acc[3] = wmma_bf16(a, b3, acc[3]);
    a = an; b0 = c0; b1 = c1; b2 = c2; b3 = c3;
  }
  acc[0] = wmma_bf16(a, b0, acc[0]);
  acc[1] = wmma_bf16(a, b1, acc[1]);
  acc[2] = wmma_bf16(a, b2, acc[2]);
  acc[3] = wmma_bf16(a, b3, acc[3]);
}

// ---------------------------------------------------------------------------
// Weight prep: f32 W[K][N] -> bf16 WT[N][K] (LDS tile transpose).
// ---------------------------------------------------------------------------
__global__ __launch_bounds__(256) void wprep_kernel(
    const float* W0, const float* W1, const float* W2, const float* W3,
    const float* W4, const float* W5, const float* W6, const float* W7,
    const float* W8, const float* W9, u16* wt) {
  __shared__ u16 tile[32][33];
  const float* src;
  switch (blockIdx.z) {
    case 0: src = W0; break; case 1: src = W1; break; case 2: src = W2; break;
    case 3: src = W3; break; case 4: src = W4; break; case 5: src = W5; break;
    case 6: src = W6; break; case 7: src = W7; break; case 8: src = W8; break;
    default: src = W9; break;
  }
  u16* dst = wt + ((size_t)blockIdx.z << 20);
  const int k0 = blockIdx.y * 32, n0 = blockIdx.x * 32;
  const int tx = threadIdx.x, ty = threadIdx.y;  // 32 x 8
#pragma unroll
  for (int i = 0; i < 4; ++i) {
    int r = ty + i * 8;
    tile[r][tx] = f2bf(src[(size_t)(k0 + r) * EMBED + n0 + tx]);
  }
  __syncthreads();
#pragma unroll
  for (int i = 0; i < 4; ++i) {
    int r = ty + i * 8;
    dst[(size_t)(n0 + r) * EMBED + k0 + tx] = tile[tx][r];
  }
}

// ---------------------------------------------------------------------------
// Activation prep: key_in -> bf16; also materialize xbf (last SEQ_Q rows/batch).
// ---------------------------------------------------------------------------
__global__ __launch_bounds__(256) void xprep_kernel(
    const float* __restrict__ key_in, u16* __restrict__ kin_bf,
    u16* __restrict__ xbf) {
  const int total = BATCH * SEQ_K * EMBED;
  for (int i = blockIdx.x * blockDim.x + threadIdx.x; i < total;
       i += gridDim.x * blockDim.x) {
    u16 h = f2bf(key_in[i]);
    kin_bf[i] = h;
    int s = (i >> 10) & 1023;
    if (s >= SEQ_Q) {
      int b = i >> 20, e = i & 1023;
      xbf[(((size_t)b * SEQ_Q + (s - SEQ_Q)) << 10) + e] = h;
    }
  }
}

// ---------------------------------------------------------------------------
// Generic WMMA GEMM: out = epilogue(A1@W1T + b1 [+ A2@W2T + b2]).
// Block = 8 waves; tile 64(M) x 128(N); wave = 16x64 strip (4 accumulators).
// MODE: 1 bf16-out | 2 RoPE->bf16 | 3 transpose->bf16 (vT) |
//       4 sigmoid*x->bf16 | 5 sigmoid->f32 | 6 final gate -> f32 (over z in place)
// ---------------------------------------------------------------------------
template <int MODE>
__global__ __launch_bounds__(256) void gemm_wmma(
    const u16* __restrict__ A1, const u16* __restrict__ W1T,
    const float* __restrict__ b1, const u16* __restrict__ A2,
    const u16* __restrict__ W2T, const float* __restrict__ b2,
    void* __restrict__ outp, const float* __restrict__ key_in,
    const int* __restrict__ key_index, const float* __restrict__ zbuf, int rpb,
    int ioff) {
  constexpr bool DUAL = (MODE >= 4);
  const int lane = threadIdx.x & 31;
  const int wv = threadIdx.x >> 5;
  const int mbase = blockIdx.y * 64 + (wv & 3) * 16;
  const int nbase = blockIdx.x * 128 + (wv >> 2) * 64;
  v8f acc[4];
#pragma unroll
  for (int t = 0; t < 4; ++t) acc[t] = zero8();

  gemm_pass(A1, W1T, acc, mbase, nbase, lane);
  if (DUAL) gemm_pass(A2, W2T, acc, mbase, nbase, lane);

  const int l = lane & 15, hi = lane >> 4;

  // RoPE: precompute per-row position indices (invariant across t).
  float idxf[8];
  if (MODE == 2) {
#pragma unroll
    for (int r = 0; r < 8; ++r) {
      int m = mbase + r + hi * 8;
      int bI = m / rpb, s = m - bI * rpb;
      idxf[r] = (float)key_index[bI * SEQ_K + ioff + s];
    }
  }

#pragma unroll
  for (int t = 0; t < 4; ++t) {
    const int n = nbase + t * 16 + l;
    float bias = b1[n];
    if (DUAL) bias += b2[n];
    float rope_inv = 0.0f;
    if (MODE == 2)
      rope_inv = __expf((float)(-2 * (n >> 1)) * (9.210340371976184f / 1024.0f));
#pragma unroll
    for (int r = 0; r < 8; ++r) {
      const int m = mbase + r + hi * 8;
      float v = acc[t][r] + bias;
      if (MODE == 2) {  // RoPE pairs (2j,2j+1) live in lanes (l, l^1), same m.
        float other = __shfl_xor(v, 1, 32);
        float sn, cs;
        __sincosf(idxf[r] * rope_inv, &sn, &cs);
        float o = (lane & 1) ? (other * sn + v * cs) : (v * cs - other * sn);
        ((u16*)outp)[((size_t)m << 10) + n] = f2bf(o);
      } else if (MODE == 3) {  // vT[b][e=n][s]
        int bI = m >> 10, s = m & 1023;
        ((u16*)outp)[(((size_t)(bI * EMBED + n)) << 10) + s] = f2bf(v);
      } else if (MODE == 1) {
        ((u16*)outp)[((size_t)m << 10) + n] = f2bf(v);
      } else if (MODE == 4) {
        float sg = 1.0f / (1.0f + __expf(-v));
        ((u16*)outp)[((size_t)m << 10) + n] = f2bf(sg * xquery(key_in, m, n));
      } else if (MODE == 5) {
        ((float*)outp)[((size_t)m << 10) + n] = 1.0f / (1.0f + __expf(-v));
      } else if (MODE == 6) {
        size_t off = ((size_t)m << 10) + n;
        float z = zbuf[off];  // z staged in d_out, read-then-overwrite in place
        float x = xquery(key_in, m, n);
        ((float*)outp)[off] = (1.0f - z) * x + z * tanhf(v);
      }
    }
  }
}

// ---------------------------------------------------------------------------
// Attention: grid (SEQ_Q/32, BATCH), block 256 (8 waves), 196 KB dynamic LDS.
// ---------------------------------------------------------------------------
#define ATTN_SMEM (32 * 1024 * 4 + 32 * 1024 * 2 + 128)

__global__ __launch_bounds__(256) void attn_kernel(
    const u16* __restrict__ q_bf, const u16* __restrict__ k_bf,
    const u16* __restrict__ vT_bf, const unsigned char* __restrict__ pad,
    u16* __restrict__ o_att, float* __restrict__ attn_out) {
  extern __shared__ char smem[];
  float* sc = (float*)smem;                                     // [32][1024] f32
  u16* pr = (u16*)(smem + 32 * 1024 * 4);                       // [32][1024] bf16
  float* invs = (float*)(smem + 32 * 1024 * 4 + 32 * 1024 * 2); // [32] 1/sum

  const int b = blockIdx.y, qt = blockIdx.x;
  const int lane = threadIdx.x & 31, wv = threadIdx.x >> 5;
  const int l = lane & 15, hi = lane >> 4;
  const u16* qb = q_bf + (size_t)b * SEQ_Q * EMBED;
  const u16* kb = k_bf + (size_t)b * SEQ_K * EMBED;
  const u16* vb = vT_bf + (size_t)b * EMBED * SEQ_K;
  const unsigned char* padb = pad + b * SEQ_K;

  // Padding bytes for this wave's 8 k-column groups (invariant across heads).
  int padv[8];
#pragma unroll
  for (int nt = 0; nt < 8; ++nt) padv[nt] = (int)padb[wv * 128 + nt * 16 + l];

  for (int h = 0; h < HEADS; ++h) {
    const int hb = h * HDIM;
    // ---- Phase 1: scores [32 q][1024 k]; wave wv owns 128 k-columns --------
    for (int mt = 0; mt < 2; ++mt) {
      // A fragments for this 16-row q strip (shared by all 8 k-tiles).
      v16bf a0 = load_frag_a(qb, qt * 32 + mt * 16, EMBED, hb, lane);
      v16bf a1 = load_frag_a(qb, qt * 32 + mt * 16, EMBED, hb + 32, lane);
      const int irow0 = SEQ_Q + qt * 32 + mt * 16 + hi * 8;  // + r below
      // Pipelined over the 8 k-tiles: prefetch next B pair before the WMMAs.
      v16bf bb0 = load_frag_b(kb, wv * 128, EMBED, hb, lane);
      v16bf bb1 = load_frag_b(kb, wv * 128, EMBED, hb + 32, lane);
      for (int nt = 0; nt < 8; ++nt) {
        v16bf nb0, nb1;
        if (nt < 7) {
          nb0 = load_frag_b(kb, wv * 128 + nt * 16 + 16, EMBED, hb, lane);
          nb1 = load_frag_b(kb, wv * 128 + nt * 16 + 16, EMBED, hb + 32, lane);
        }
        v8f acc = zero8();
        acc = wmma_bf16(a0, bb0, acc);
        acc = wmma_bf16(a1, bb1, acc);
        const int kg = wv * 128 + nt * 16 + l;
        float* scol = sc + kg + (mt * 16 + hi * 8) * 1024;
#pragma unroll
        for (int r = 0; r < 8; ++r) {
          int irow = irow0 + r;
          int masked = (((kg > irow) ? 1 : 0) | padv[nt]) & ((kg != irow) ? 1 : 0);
          scol[r * 1024] = masked ? -1e30f : acc[r] * 0.125f;  // 1/sqrt(64)
        }
        bb0 = nb0; bb1 = nb1;
      }
    }
    __syncthreads();

    // ---- Phase 2: softmax rows; wave wv owns rows 4*wv..4*wv+3 -------------
    for (int rr = 0; rr < 4; ++rr) {
      const int row = wv * 4 + rr;
      float* srow = sc + row * 1024;
      u16* prow = pr + row * 1024;
      float mx = -1e30f;
      for (int i = lane; i < SEQ_K; i += 32) mx = fmaxf(mx, srow[i]);
#pragma unroll
      for (int off = 16; off; off >>= 1) mx = fmaxf(mx, __shfl_xor(mx, off, 32));
      float sum = 0.0f;
      for (int i = lane; i < SEQ_K; i += 32) {
        float e = __expf(srow[i] - mx);
        prow[i] = f2bf(e);
        sum += e;
      }
#pragma unroll
      for (int off = 16; off; off >>= 1) sum += __shfl_xor(sum, off, 32);
      float inv = 1.0f / sum;
      if (lane == 0) invs[row] = inv;
      // mean over heads into d_out attn half (block owns rows: no atomics).
      float sc16 = inv * (1.0f / (float)HEADS);
      float* arow = attn_out + ((size_t)b * SEQ_Q + qt * 32 + row) * SEQ_K;
      if (h == 0) {
        for (int i = lane; i < SEQ_K; i += 32) arow[i] = bf2f(prow[i]) * sc16;
      } else {
        for (int i = lane; i < SEQ_K; i += 32) arow[i] += bf2f(prow[i]) * sc16;
      }
    }
    __syncthreads();

    // ---- Phase 3: O[32][64] = probs @ V ; wave wv -> (mt=wv&1, nt=wv>>1) ---
    {
      const int mt = wv & 1, nt = wv >> 1;
      v8f acc = zero8();
      // Pipelined: A from LDS probs, B from global vT; prefetch next pair.
      v16bf a = load_frag_a(pr, mt * 16, 1024, 0, lane);
      v16bf bb = load_frag_b(vb, hb + nt * 16, SEQ_K, 0, lane);
      for (int kc = 0; kc < SEQ_K - 32; kc += 32) {
        v16bf an = load_frag_a(pr, mt * 16, 1024, kc + 32, lane);
        v16bf bn = load_frag_b(vb, hb + nt * 16, SEQ_K, kc + 32, lane);
        acc = wmma_bf16(a, bb, acc);
        a = an; bb = bn;
      }
      acc = wmma_bf16(a, bb, acc);
      const int col = hb + nt * 16 + l;
      u16* orow = o_att + ((size_t)b * SEQ_Q + qt * 32 + mt * 16 + hi * 8) * EMBED;
#pragma unroll
      for (int r = 0; r < 8; ++r) {
        float val = acc[r] * invs[mt * 16 + r + hi * 8];
        orow[(size_t)r * EMBED + col] = f2bf(val);
      }
    }
    __syncthreads();  // protect sc/pr/invs before next head
  }
}

// ---------------------------------------------------------------------------
extern "C" void kernel_launch(void* const* d_in, const int* in_sizes, int n_in,
                              void* d_out, int out_size, void* d_ws, size_t ws_size,
                              hipStream_t stream) {
  (void)in_sizes; (void)n_in; (void)out_size; (void)ws_size;
  const float* key_in = (const float*)d_in[0];
  const int* key_index = (const int*)d_in[1];
  const unsigned char* key_pad = (const unsigned char*)d_in[2];  // jnp bool = 1B
  const float* Wq = (const float*)d_in[3];  const float* bq = (const float*)d_in[4];
  const float* Wk = (const float*)d_in[5];  const float* bk = (const float*)d_in[6];
  const float* Wv = (const float*)d_in[7];  const float* bv = (const float*)d_in[8];
  const float* Wo = (const float*)d_in[9];  const float* bo = (const float*)d_in[10];
  const float* Wxr = (const float*)d_in[11]; const float* bxr = (const float*)d_in[12];
  const float* Wyr = (const float*)d_in[13]; const float* byr = (const float*)d_in[14];
  const float* Wxz = (const float*)d_in[15]; const float* bxz = (const float*)d_in[16];
  const float* Wyz = (const float*)d_in[17]; const float* byz = (const float*)d_in[18];
  const float* Wxg = (const float*)d_in[19]; const float* bxg = (const float*)d_in[20];
  const float* Wyg = (const float*)d_in[21]; const float* byg = (const float*)d_in[22];

  // Workspace carving (bytes); all 1 MB-aligned offsets. Total ~189 MB.
  char* ws = (char*)d_ws;
  const size_t WSZ = (size_t)EMBED * EMBED * 2;        // 2 MB per bf16 weight
  u16* WT = (u16*)ws;                                  // 10 matrices, 20 MB
  u16* kin_bf = (u16*)(ws + 10 * WSZ);                 // 33.5 MB
  u16* xbf = (u16*)(ws + 10 * WSZ + 33554432);         // 16.7 MB
  u16* q_bf = (u16*)(ws + 10 * WSZ + 50331648);        // 16.7 MB (reused as rx_bf)
  u16* k_bf = (u16*)(ws + 10 * WSZ + 67108864);        // 33.5 MB
  u16* vT_bf = (u16*)(ws + 10 * WSZ + 100663296);      // 33.5 MB
  u16* o_att = (u16*)(ws + 10 * WSZ + 134217728);      // 16.7 MB
  u16* oproj_bf = (u16*)(ws + 10 * WSZ + 150994944);   // 16.7 MB
  u16* rx_bf = q_bf;                                   // q dead after attention
  float* out_f = (float*)d_out;                        // [16,512,1024] final out
  float* attn_out = out_f + (size_t)BATCH * SEQ_Q * EMBED;  // [16,512,1024]
  float* zbuf = out_f;                                 // z staged in-place

  u16* WTq = WT + 0 * (WSZ / 2); u16* WTk = WT + 1 * (WSZ / 2);
  u16* WTv = WT + 2 * (WSZ / 2); u16* WTo = WT + 3 * (WSZ / 2);
  u16* WTxr = WT + 4 * (WSZ / 2); u16* WTyr = WT + 5 * (WSZ / 2);
  u16* WTxz = WT + 6 * (WSZ / 2); u16* WTyz = WT + 7 * (WSZ / 2);
  u16* WTxg = WT + 8 * (WSZ / 2); u16* WTyg = WT + 9 * (WSZ / 2);

  wprep_kernel<<<dim3(32, 32, 10), dim3(32, 8), 0, stream>>>(
      Wq, Wk, Wv, Wo, Wxr, Wyr, Wxz, Wyz, Wxg, Wyg, WT);
  xprep_kernel<<<4096, 256, 0, stream>>>(key_in, kin_bf, xbf);

  const u16* NUL = nullptr;
  // q = rope(x@Wq + bq)
  gemm_wmma<2><<<dim3(8, SEQ_Q * BATCH / 64), 256, 0, stream>>>(
      xbf, WTq, bq, NUL, NUL, nullptr, q_bf, key_in, key_index, nullptr, SEQ_Q, SEQ_Q);
  // k = rope(kin@Wk + bk)
  gemm_wmma<2><<<dim3(8, SEQ_K * BATCH / 64), 256, 0, stream>>>(
      kin_bf, WTk, bk, NUL, NUL, nullptr, k_bf, key_in, key_index, nullptr, SEQ_K, 0);
  // vT = transpose(kin@Wv + bv)
  gemm_wmma<3><<<dim3(8, SEQ_K * BATCH / 64), 256, 0, stream>>>(
      kin_bf, WTv, bv, NUL, NUL, nullptr, vT_bf, key_in, key_index, nullptr, 0, 0);
  // attention (also writes mean attn weights to d_out second half)
  attn_kernel<<<dim3(SEQ_Q / 32, BATCH), 256, ATTN_SMEM, stream>>>(
      q_bf, k_bf, vT_bf, key_pad, o_att, attn_out);
  // oproj = o_att@Wo + bo (bf16)
  gemm_wmma<1><<<dim3(8, SEQ_Q * BATCH / 64), 256, 0, stream>>>(
      o_att, WTo, bo, NUL, NUL, nullptr, oproj_bf, key_in, key_index, nullptr, 0, 0);
  // rx = sigmoid(x@Wxr + o@Wyr + bxr + byr) * x
  gemm_wmma<4><<<dim3(8, SEQ_Q * BATCH / 64), 256, 0, stream>>>(
      xbf, WTxr, bxr, oproj_bf, WTyr, byr, rx_bf, key_in, key_index, nullptr, 0, 0);
  // z = sigmoid(x@Wxz + o@Wyz + bxz + byz) staged into d_out
  gemm_wmma<5><<<dim3(8, SEQ_Q * BATCH / 64), 256, 0, stream>>>(
      xbf, WTxz, bxz, oproj_bf, WTyz, byz, zbuf, key_in, key_index, nullptr, 0, 0);
  // out = (1-z)*x + z*tanh(rx@Wxg + o@Wyg + bxg + byg), overwrites z in place
  gemm_wmma<6><<<dim3(8, SEQ_Q * BATCH / 64), 256, 0, stream>>>(
      rx_bf, WTxg, bxg, oproj_bf, WTyg, byg, out_f, key_in, key_index, zbuf, 0, 0);
}
